// Attention_1743756722801
// MI455X (gfx1250) — matrix operand order repaired and checked by reading, not compiled, not run on hardware
//
#include <hip/hip_runtime.h>
#include <hip/hip_bf16.h>
#include <stdint.h>

// ---------------- problem constants ----------------
#define BATCH   8
#define NTOK    1024
#define EMB     1024
#define NH      16
#define HD      64          // head dim; scale = 1/8
#define D3      (3*EMB)     // 3072

typedef __bf16 v16bf __attribute__((ext_vector_type(16)));
typedef __bf16 v8bf  __attribute__((ext_vector_type(8)));
typedef float  v8f   __attribute__((ext_vector_type(8)));

static __device__ inline v8f vzero8() {
    v8f z;
#pragma unroll
    for (int j = 0; j < 8; ++j) z[j] = 0.0f;
    return z;
}

// Build a 16-bit A-fragment (16x32, MxK) from two 8-element chunks.
// Callers use contiguous k: lane L holds M = L&15; elements 0..7 = K = (L>>4)*16 + {0..7},
// elements 8..15 = K = (L>>4)*16 + 8 + {0..7}; B frags use the same order.
static __device__ inline v16bf make_afrag(const __bf16* lo_p, const __bf16* hi_p) {
    v8bf lo = *(const v8bf*)lo_p;
    v8bf hi = *(const v8bf*)hi_p;
    v16bf a;
#pragma unroll
    for (int i = 0; i < 8; ++i) { a[i] = lo[i]; a[i + 8] = hi[i]; }
    return a;
}

// CDNA5 async global->LDS copy (ASYNCcnt path), 16 bytes per lane.
// low 32 bits of a generic __shared__
// pointer are the workgroup-relative LDS byte offset.
static __device__ inline void async_b128(void* lds_dst, const void* gsrc) {
    uint32_t loff = (uint32_t)(uintptr_t)lds_dst;
    asm volatile("global_load_async_to_lds_b128 %0, %1, off"
                 :: "v"(loff), "v"(gsrc) : "memory");
}
static __device__ inline void wait_async0() {
    asm volatile("s_wait_asynccnt 0x0" ::: "memory");
}

// ---------------- kernel 1: fp32 -> bf16 convert ----------------
__global__ void cvt_f32_bf16_kernel(const float* __restrict__ src,
                                    __bf16* __restrict__ dst, int n) {
    int i = blockIdx.x * blockDim.x + threadIdx.x;
    if (i < n) dst[i] = (__bf16)src[i];
}

// ---------------- kernel 2/5: bf16 WMMA GEMM  out = A @ W^T (+bias) ------
// A: M x K row-major bf16.  W: Nout x K row-major bf16.
// Block: 256 threads = 8 waves; block tile 128 x 64; wave tile 16 x 64.
// B tile (64 cols x 32 k = 4KB) staged in LDS via async loads, double
// buffered, shared by all 8 waves. Prefetch issue is unconditional with a
// wrapped offset (last issue refills the dead buffer; s_endpgm waits idle).
template <bool F32OUT, bool HAS_BIAS>
__global__ __launch_bounds__(256)
void gemm_bf16_kernel(const __bf16* __restrict__ A, const __bf16* __restrict__ W,
                      float* __restrict__ outF, __bf16* __restrict__ outB,
                      const float* __restrict__ bias, int M, int Nout, int K) {
    __shared__ __align__(16) __bf16 bsh[2][64 * 32];   // 2 x 4KB

    const int nblocks = Nout / 64;
    const int bn = blockIdx.x % nblocks;
    const int bm = blockIdx.x / nblocks;
    const int wave = threadIdx.x >> 5;
    const int lane = threadIdx.x & 31;
    const int l15 = lane & 15;
    const int h16 = lane >> 4;

    const int m0 = bm * 128 + wave * 16;
    const int n0 = bn * 64;

    // async-copy slot: thread t moves the 16B chunk (col = t>>2, part = t&3)
    const int cc = threadIdx.x >> 2;   // 0..63 : B column
    const int cp = threadIdx.x & 3;    // 0..3  : 16B part of the 64B k-row
    const __bf16* wsrc = W + (size_t)(n0 + cc) * K + cp * 8;
    const int lslot = cc * 32 + cp * 8;

    v8f acc[4];
#pragma unroll
    for (int i = 0; i < 4; ++i) acc[i] = vzero8();

    const __bf16* arow = A + (size_t)(m0 + l15) * K;

    async_b128(&bsh[0][lslot], wsrc);               // tile 0
    const int nk = K / 32;
    for (int i = 0; i < nk; ++i) {
        const int k0 = i * 32;
        wait_async0();                               // own tile-i chunk done
        __syncthreads();                             // all chunks visible
        const int knext = (i + 1 < nk) ? (k0 + 32) : 0;   // wrap, no branch
        async_b128(&bsh[(i + 1) & 1][lslot], wsrc + knext);

        v16bf a = make_afrag(arow + k0 + h16 * 16, arow + k0 + h16 * 16 + 8);
        const __bf16* bt = &bsh[i & 1][0];
        v16bf bfr[4];
#pragma unroll
        for (int nt = 0; nt < 4; ++nt)
            bfr[nt] = *(const v16bf*)(bt + (nt * 16 + l15) * 32 + h16 * 16);
#pragma unroll
        for (int nt = 0; nt < 4; ++nt)
            acc[nt] = __builtin_amdgcn_wmma_f32_16x16x32_bf16(
                false, a, false, bfr[nt], (short)0, acc[nt], false, false);
    }

    // epilogue: C/D layout -> row m = r + 8*h16, col = n0 + nt*16 + l15
#pragma unroll
    for (int nt = 0; nt < 4; ++nt) {
        const int col = n0 + nt * 16 + l15;
        float bv = 0.0f;
        if constexpr (HAS_BIAS) bv = bias[col];
#pragma unroll
        for (int r = 0; r < 8; ++r) {
            const int row = m0 + r + 8 * h16;
            const float v = acc[nt][r] + bv;
            if constexpr (F32OUT) outF[(size_t)row * Nout + col] = v;
            else                  outB[(size_t)row * Nout + col] = (__bf16)v;
        }
    }
}

// ---------------- kernel 3: RoPE-2D + head repack ----------------
__global__ __launch_bounds__(64)
void rope_pack_kernel(const __bf16* __restrict__ qkv,
                      const long long* __restrict__ xpos,
                      __bf16* __restrict__ Qb, __bf16* __restrict__ Kb,
                      __bf16* __restrict__ Vtb) {
    const int bn = blockIdx.x;            // b*N + n
    const int b = bn >> 10;
    const int n = bn & 1023;
    const int t = threadIdx.x;            // 0..63 (head dim)
    const int half = t >> 5;              // which rope half / pos index
    const int loc = t & 31;
    const int j = loc & 15;

    const float pos = (float)xpos[(size_t)bn * 2 + half];
    const float invf = powf(100.0f, -(float)j * (1.0f / 16.0f));
    const float ang = pos * invf;
    const float cs = cosf(ang), sn = sinf(ang);
    const float sgn = (loc < 16) ? -1.0f : 1.0f;

    const __bf16* row = qkv + (size_t)bn * D3;

    for (int h = 0; h < NH; ++h) {
        const float qv = (float)row[h * HD + t];
        const float kv = (float)row[EMB + h * HD + t];
        const float vv = (float)row[2 * EMB + h * HD + t];
        const float qp = __shfl_xor(qv, 16, 32);   // rope partner (t ^ 16)
        const float kp = __shfl_xor(kv, 16, 32);
        const float qo = (qv * cs + sgn * qp * sn) * 0.125f;  // fold 1/sqrt(hd)
        const float ko = (kv * cs + sgn * kp * sn);
        const size_t bh = (size_t)(b * NH + h);
        Qb[(bh * NTOK + n) * HD + t] = (__bf16)qo;
        Kb[(bh * NTOK + n) * HD + t] = (__bf16)ko;
        Vtb[(bh * HD + t) * NTOK + n] = (__bf16)vv;   // transposed store
    }
}

// ---------------- kernel 4: flash attention ----------------
// Block = 256 threads = 8 waves over one (b,h); each wave owns a 16-query
// tile. The shared 32-key K tile (32x64) and V tile (64x32, from V^T) are
// staged in LDS with async loads (double buffered), then every wave does
// S = Q K^T (4 wmma), online softmax, O += P V (4 wmma) per step.
__global__ __launch_bounds__(256)
void flash_attn_kernel(const __bf16* __restrict__ Qb, const __bf16* __restrict__ Kb,
                       const __bf16* __restrict__ Vtb, __bf16* __restrict__ Ob) {
    __shared__ __align__(16) __bf16 ksh[2][32 * 64];   // [key][d]   2 x 4KB
    __shared__ __align__(16) __bf16 vsh[2][64 * 32];   // [d][k]     2 x 4KB
    __shared__ __align__(16) __bf16 pbuf[8 * 16 * 32]; // P bounce   8KB

    const int wave = threadIdx.x >> 5;
    const int lane = threadIdx.x & 31;
    const int l15 = lane & 15;
    const int h16 = lane >> 4;

    const int tiles128 = NTOK / 128;                   // 8
    const int bh = blockIdx.x / tiles128;
    const int t128 = blockIdx.x % tiles128;
    const int q0 = t128 * 128 + wave * 16;

    const __bf16* Q  = Qb  + (size_t)bh * NTOK * HD;
    const __bf16* Kp = Kb  + (size_t)bh * NTOK * HD;
    const __bf16* Vt = Vtb + (size_t)bh * HD * NTOK;

    // async-copy slots: K tile 32 rows x 128B, V tile 64 rows x 64B
    const int krow = threadIdx.x >> 3, kpart = threadIdx.x & 7;  // 32 x 8
    const int vrow = threadIdx.x >> 2, vpart = threadIdx.x & 3;  // 64 x 4
    const int kslot = krow * 64 + kpart * 8;
    const int vslot = vrow * 32 + vpart * 8;
    const __bf16* ksrc = Kp + (size_t)krow * HD + kpart * 8;   // + kt*HD
    const __bf16* vsrc = Vt + (size_t)vrow * NTOK + vpart * 8; // + kt

    // Q fragments (hd=64 -> 2 A-frags of 16x32), kept in registers
    v16bf aq[2];
    const __bf16* qrow = Q + (size_t)(q0 + l15) * HD;
#pragma unroll
    for (int f = 0; f < 2; ++f)
        aq[f] = make_afrag(qrow + f * 32 + h16 * 16, qrow + f * 32 + h16 * 16 + 8);

    float mrow[8], lrow[8];
    v8f acc[4];
#pragma unroll
    for (int r = 0; r < 8; ++r) { mrow[r] = -1e30f; lrow[r] = 0.0f; }
#pragma unroll
    for (int i = 0; i < 4; ++i) acc[i] = vzero8();

    __bf16* pb = pbuf + wave * (16 * 32);

    // prologue: stage key-tile 0
    async_b128(&ksh[0][kslot], ksrc);
    async_b128(&vsh[0][vslot], vsrc);

    const int nkt = NTOK / 32;                         // 32 steps
    for (int i = 0; i < nkt; ++i) {
        const int kt = i * 32;
        wait_async0();
        __syncthreads();
        const int ktn = (i + 1 < nkt) ? (kt + 32) : 0;  // wrap, no branch
        async_b128(&ksh[(i + 1) & 1][kslot], ksrc + (size_t)ktn * HD);
        async_b128(&vsh[(i + 1) & 1][vslot], vsrc + ktn);

        const __bf16* kt_sh = &ksh[i & 1][0];
        const __bf16* vt_sh = &vsh[i & 1][0];

        // S = Q K^T : load all 4 B-frags, then 4 back-to-back wmma
        v16bf bk[4];
#pragma unroll
        for (int f = 0; f < 2; ++f) {
            bk[f * 2]     = *(const v16bf*)(kt_sh + l15 * HD + f * 32 + h16 * 16);
            bk[f * 2 + 1] = *(const v16bf*)(kt_sh + (16 + l15) * HD + f * 32 + h16 * 16);
        }
        v8f c0 = vzero8(), c1 = vzero8();
        c0 = __builtin_amdgcn_wmma_f32_16x16x32_bf16(false, aq[0], false, bk[0], (short)0, c0, false, false);
        c1 = __builtin_amdgcn_wmma_f32_16x16x32_bf16(false, aq[0], false, bk[1], (short)0, c1, false, false);
        c0 = __builtin_amdgcn_wmma_f32_16x16x32_bf16(false, aq[1], false, bk[2], (short)0, c0, false, false);
        c1 = __builtin_amdgcn_wmma_f32_16x16x32_bf16(false, aq[1], false, bk[3], (short)0, c1, false, false);

        // online softmax; row m = r + 8*h16 lives in a 16-lane half, so
        // xor-reduce with masks 8,4,2,1 stays in-group.
        float alpha[8];
#pragma unroll
        for (int r = 0; r < 8; ++r) {
            float v = fmaxf(c0[r], c1[r]);
#pragma unroll
            for (int off = 8; off > 0; off >>= 1) v = fmaxf(v, __shfl_xor(v, off, 32));
            const float nm = fmaxf(mrow[r], v);
            alpha[r] = expf(mrow[r] - nm);
            mrow[r] = nm;
            c0[r] = expf(c0[r] - nm);
            c1[r] = expf(c1[r] - nm);
            float s = c0[r] + c1[r];
#pragma unroll
            for (int off = 8; off > 0; off >>= 1) s += __shfl_xor(s, off, 32);
            lrow[r] = lrow[r] * alpha[r] + s;
        }
#pragma unroll
        for (int nt = 0; nt < 4; ++nt)
#pragma unroll
            for (int r = 0; r < 8; ++r) acc[nt][r] *= alpha[r];

        // P tile (16x32) to per-wave LDS: C-layout -> A-layout bounce
#pragma unroll
        for (int r = 0; r < 8; ++r) {
            const int m = r + 8 * h16;
            pb[m * 32 + l15]      = (__bf16)c0[r];
            pb[m * 32 + 16 + l15] = (__bf16)c1[r];
        }
        asm volatile("s_wait_dscnt 0x0" ::: "memory");  // in-wave DS store->load

        const __bf16* prow = pb + l15 * 32;
        v16bf ap = make_afrag(prow + h16 * 16, prow + h16 * 16 + 8);

        // O += P @ V : load all 4 B-frags, then 4 back-to-back wmma
        v16bf bv[4];
#pragma unroll
        for (int nt = 0; nt < 4; ++nt)
            bv[nt] = *(const v16bf*)(vt_sh + (nt * 16 + l15) * 32 + h16 * 16);
#pragma unroll
        for (int nt = 0; nt < 4; ++nt)
            acc[nt] = __builtin_amdgcn_wmma_f32_16x16x32_bf16(
                false, ap, false, bv[nt], (short)0, acc[nt], false, false);
    }

    // epilogue: O/l, write to (b, n, h*64+d) bf16 for the projection GEMM
    const int b = bh / NH, h = bh % NH;
#pragma unroll
    for (int nt = 0; nt < 4; ++nt)
#pragma unroll
        for (int r = 0; r < 8; ++r) {
            const int m = r + 8 * h16;
            const float v = acc[nt][r] / lrow[r];
            Ob[((size_t)(b * NTOK + q0 + m)) * EMB + h * HD + nt * 16 + l15] = (__bf16)v;
        }
}

// ---------------- launcher ----------------
extern "C" void kernel_launch(void* const* d_in, const int* in_sizes, int n_in,
                              void* d_out, int out_size, void* d_ws, size_t ws_size,
                              hipStream_t stream) {
    const float*      x      = (const float*)d_in[0];      // (8,1024,1024)
    const long long*  xpos   = (const long long*)d_in[1];  // (8,1024,2) int64
    const float*      w_qkv  = (const float*)d_in[2];      // (3072,1024)
    const float*      w_proj = (const float*)d_in[3];      // (1024,1024)
    const float*      b_proj = (const float*)d_in[4];      // (1024,)
    float*            out    = (float*)d_out;              // (8,1024,1024)

    char* ws = (char*)d_ws;
    size_t off = 0;
    __bf16* xb     = (__bf16*)(ws + off); off += (size_t)BATCH * NTOK * EMB * 2;     // 16 MB
    __bf16* wqkvb  = (__bf16*)(ws + off); off += (size_t)D3 * EMB * 2;               //  6 MB
    __bf16* wprojb = (__bf16*)(ws + off); off += (size_t)EMB * EMB * 2;              //  2 MB
    __bf16* qkvb   = (__bf16*)(ws + off); off += (size_t)BATCH * NTOK * D3 * 2;      // 48 MB
    __bf16* Qb     = (__bf16*)(ws + off); off += (size_t)BATCH * NH * NTOK * HD * 2; // 16 MB
    __bf16* Kb     = (__bf16*)(ws + off); off += (size_t)BATCH * NH * NTOK * HD * 2; // 16 MB
    __bf16* Vtb    = (__bf16*)(ws + off); off += (size_t)BATCH * NH * HD * NTOK * 2; // 16 MB
    __bf16* attnb  = xb;   // alias: x's bf16 copy is dead after the QKV GEMM

    const int nx  = BATCH * NTOK * EMB;     // 8388608
    const int nwq = D3 * EMB;               // 3145728
    const int nwp = EMB * EMB;              // 1048576

    cvt_f32_bf16_kernel<<<(nx  + 255) / 256, 256, 0, stream>>>(x, xb, nx);
    cvt_f32_bf16_kernel<<<(nwq + 255) / 256, 256, 0, stream>>>(w_qkv, wqkvb, nwq);
    cvt_f32_bf16_kernel<<<(nwp + 255) / 256, 256, 0, stream>>>(w_proj, wprojb, nwp);

    // qkv = x @ w_qkv^T : M=8192, Nout=3072, K=1024 -> bf16 out, no bias
    {
        const int M = BATCH * NTOK, Nout = D3, K = EMB;
        dim3 grid((Nout / 64) * (M / 128));
        gemm_bf16_kernel<false, false><<<grid, 256, 0, stream>>>(
            xb, wqkvb, nullptr, qkvb, nullptr, M, Nout, K);
    }

    rope_pack_kernel<<<BATCH * NTOK, 64, 0, stream>>>(qkvb, xpos, Qb, Kb, Vtb);

    flash_attn_kernel<<<BATCH * NH * (NTOK / 128), 256, 0, stream>>>(Qb, Kb, Vtb, attnb);

    // out = attn @ w_proj^T + b : M=8192, Nout=1024, K=1024 -> f32 out, bias
    {
        const int M = BATCH * NTOK, Nout = EMB, K = EMB;
        dim3 grid((Nout / 64) * (M / 128));
        gemm_bf16_kernel<true, true><<<grid, 256, 0, stream>>>(
            attnb, wprojb, out, nullptr, b_proj, M, Nout, K);
    }
}
